// ADvectorComplex_12472585027843
// MI455X (gfx1250) — matrix operand order repaired and checked by reading, not compiled, not run on hardware
//
#include <hip/hip_runtime.h>
#include <math.h>

// ---------------- problem dims ----------------
constexpr int NB = 32, NT = 1024, NI = 256, NH = 256;
constexpr size_t BT      = (size_t)NB * NT;     // 32768 flattened (b,t) rows
constexpr size_t BTH     = BT * NH;             // 8,388,608
constexpr size_t PLANE_E = 256 * 256;           // one weight plane (elements)

// ---------------- workspace layout (bytes) ----------------
// [W planes]  9 bf16 planes: Wih{R,I,S}, Whh{R,I,S}, We{R,I,S}
// [xw]        3 x (T,B,H) complex fp32   (scan additive input, t-major)
// [hs]        3 x 3 bf16 planes (B*T,H)  (R, I, R+I for kernel-3 A-matrix)
// [emb]       3 x (B*T,I) complex fp32
// [sc]        3 x (B*T) fp32
constexpr size_t SZ_W    = 9 * PLANE_E * 2;
constexpr size_t OFF_XW  = SZ_W;
constexpr size_t SZ_XW1  = BTH * 8;
constexpr size_t OFF_HS  = OFF_XW + 3 * SZ_XW1;
constexpr size_t SZ_HSP  = BTH * 2;
constexpr size_t OFF_EMB = OFF_HS + 9 * SZ_HSP;
constexpr size_t SZ_EMB1 = BT * NI * 8;
constexpr size_t OFF_SC  = OFF_EMB + 3 * SZ_EMB1;

typedef __attribute__((ext_vector_type(16))) __bf16       v16bf;
typedef __attribute__((ext_vector_type(8)))  float        v8f;
typedef __attribute__((ext_vector_type(4)))  unsigned int uv4;
typedef __attribute__((ext_vector_type(4)))  unsigned int v4u;
typedef __attribute__((ext_vector_type(4)))  int          v4i;
typedef __attribute__((ext_vector_type(8)))  int          v8i;

struct cplx { float x, y; };

// complex tanh: tanh(x+iy) = (sinh 2x + i sin 2y) / (cosh 2x + cos 2y)
__device__ inline cplx ctanhz(float x, float y) {
    float x2 = 2.f * x, y2 = 2.f * y;
    cplx r;
    if (fabsf(x2) > 30.f) { r.x = (x2 > 0.f) ? 1.f : -1.f; r.y = 0.f; return r; }
    float d = coshf(x2) + cosf(y2);
    r.x = sinhf(x2) / d;
    r.y = sinf(y2) / d;
    return r;
}

__device__ inline v8f bwmma(v16bf a, v16bf b, v8f c) {
    return __builtin_amdgcn_wmma_f32_16x16x32_bf16(false, a, false, b, (short)0, c, false, false);
}

union FragU { v16bf v; uv4 q[2]; };

// A fragment (16x32 bf16), source row-major with K-stride 256.
// lane: m = lane&15 ; K = grp*8 + {0..7}  and  16 + grp*8 + {0..7}  (grp = lane>>4)
__device__ inline v16bf fragA(const __bf16* __restrict__ p, size_t row_base, int lane, int kk) {
    const __bf16* r = p + (row_base + (size_t)(lane & 15)) * 256 + kk + (lane >> 4) * 8;
    FragU u;
    u.q[0] = *(const uv4*)(r);
    u.q[1] = *(const uv4*)(r + 16);
    return u.v;
}

// B fragment (32x16 bf16) of C = A * W^T : B[k][n] = W[n][k], W row-major [N][256].
// lane: n = lane&15 ; K = grp*8 + {0..7}  and  16 + grp*8 + {0..7}  (grp = lane>>4)
__device__ inline v16bf fragB(const __bf16* __restrict__ p, int n_base, int lane, int kk) {
    const __bf16* r = p + (size_t)(n_base + (lane & 15)) * 256 + kk + (lane >> 4) * 8;
    FragU u;
    u.q[0] = *(const uv4*)(r);
    u.q[1] = *(const uv4*)(r + 16);
    return u.v;
}

// ---------------- prep: weights -> bf16 R / I / (R+I) planes ----------------
__global__ __launch_bounds__(256) void k_prep(const float* __restrict__ Wih,
                                              const float* __restrict__ Whh,
                                              const float* __restrict__ We,
                                              __bf16* __restrict__ planes) {
    int gid = blockIdx.x * 256 + threadIdx.x;           // 3 * 65536 total
    int mat = gid >> 16, e = gid & 65535;
    const float* src = (mat == 0) ? Wih : (mat == 1 ? Whh : We);
    float re = src[2 * e], im = src[2 * e + 1];
    __bf16* base = planes + (size_t)mat * 3 * PLANE_E;
    base[e]               = (__bf16)re;
    base[PLANE_E + e]     = (__bf16)im;
    base[2 * PLANE_E + e] = (__bf16)(re + im);
}

// ---------------- k1: xw = x * Wih^T + (bih+bhh) ----------------
// One block per 16-row m-tile; TDM stages the 32 KB contiguous fp32 tile into LDS,
// 8 waves each compute two 16x16 complex output tiles with Gauss 3-mult WMMA.
__global__ __launch_bounds__(256) void k1_inproj(
    const float* __restrict__ x, const float* __restrict__ bih, const float* __restrict__ bhh,
    const __bf16* __restrict__ WR, const __bf16* __restrict__ WI, const __bf16* __restrict__ WS,
    float* __restrict__ xw) {
    __shared__ __attribute__((aligned(16))) float  sraw[16 * 512];   // 16 rows x 256 complex
    __shared__ __attribute__((aligned(16))) __bf16 aR[16 * 256], aI[16 * 256], aS[16 * 256];

    const int tid = threadIdx.x, lane = tid & 31, wv = tid >> 5;
    const int mt = blockIdx.x;

    if (wv == 0) {  // wave 0 drives the Tensor Data Mover (EXEC-independent issue)
        const unsigned long long gsrc = (unsigned long long)(uintptr_t)(x + (size_t)mt * 16 * 512);
        const unsigned ldsa = (unsigned)(uintptr_t)(void*)sraw;
        v4u g0;
        g0.x = 1u;                                       // count=1 valid descriptor
        g0.y = ldsa;                                     // lds_addr
        g0.z = (unsigned)gsrc;                           // global_addr lo
        g0.w = (unsigned)((gsrc >> 32) & 0x1FFFFFFu) | (2u << 30);  // addr hi | type=2
        v8i g1;
        g1[0] = (int)(2u << 16);                         // data_size = 4B
        g1[1] = (int)(8192u << 16);                      // tensor_dim0 = 8192 (bits 63:48)
        g1[2] = 0;                                       // dim0 hi / dim1 lo
        g1[3] = (int)(8192u << 16);                      // tile_dim0 = 8192 (bits 127:112)
        g1[4] = 0;                                       // tile_dim1/2 unused -> 1D tile
        g1[5] = 8192;                                    // tensor_dim0_stride lo
        g1[6] = 0; g1[7] = 0;
        v4i z4 = {0, 0, 0, 0};
#if __clang_major__ >= 23
        v8i z8 = {0, 0, 0, 0, 0, 0, 0, 0};
        __builtin_amdgcn_tensor_load_to_lds(g0, g1, z4, z4, z8, 0);
#else
        __builtin_amdgcn_tensor_load_to_lds(g0, g1, z4, z4, 0);
#endif
        __builtin_amdgcn_s_wait_tensorcnt(0);
    }
    __syncthreads();

    for (int e = tid; e < 4096; e += 256) {              // fp32 complex -> bf16 planes
        float re = sraw[2 * e], im = sraw[2 * e + 1];
        aR[e] = (__bf16)re; aI[e] = (__bf16)im; aS[e] = (__bf16)(re + im);
    }
    __syncthreads();

    const int nt0 = wv * 2;
    v8f t1a = {0,0,0,0,0,0,0,0}, t2a = t1a, t3a = t1a, t1b = t1a, t2b = t1a, t3b = t1a;
#pragma unroll
    for (int kk = 0; kk < 256; kk += 32) {
        v16bf ar = fragA(aR, 0, lane, kk);
        v16bf ai = fragA(aI, 0, lane, kk);
        v16bf as = fragA(aS, 0, lane, kk);
        v16bf b0r = fragB(WR, nt0 * 16, lane, kk);
        v16bf b0i = fragB(WI, nt0 * 16, lane, kk);
        v16bf b0s = fragB(WS, nt0 * 16, lane, kk);
        t1a = bwmma(ar, b0r, t1a); t2a = bwmma(ai, b0i, t2a); t3a = bwmma(as, b0s, t3a);
        v16bf b1r = fragB(WR, nt0 * 16 + 16, lane, kk);
        v16bf b1i = fragB(WI, nt0 * 16 + 16, lane, kk);
        v16bf b1s = fragB(WS, nt0 * 16 + 16, lane, kk);
        t1b = bwmma(ar, b1r, t1b); t2b = bwmma(ai, b1i, t2b); t3b = bwmma(as, b1s, t3b);
    }

    const int grp = lane >> 4, col = lane & 15;
#pragma unroll
    for (int s = 0; s < 2; ++s) {
        int h = (nt0 + s) * 16 + col;
        float br_ = bih[2 * h] + bhh[2 * h];
        float bi_ = bih[2 * h + 1] + bhh[2 * h + 1];
        v8f T1 = s ? t1b : t1a;
        v8f T2 = s ? t2b : t2a;
        v8f T3 = s ? t3b : t3a;
#pragma unroll
        for (int r = 0; r < 8; ++r) {
            int m = mt * 16 + r + 8 * grp;
            int b = m >> 10, t = m & 1023;               // row-major (b,t) flatten, T=1024
            float cr = T1[r] - T2[r] + br_;
            float ci = T3[r] - T1[r] - T2[r] + bi_;
            float* dst = xw + (((size_t)t * NB + b) * NH + h) * 2;  // (t,b,h) for the scan
            dst[0] = cr; dst[1] = ci;
        }
    }
}

// ---------------- k2: sequential scan h_t = ctanh(xw_t + h * Whh^T) ----------------
// grid = 3 (one workgroup per input), 1024 threads = 32 waves; wave -> (mt,nt) tile.
// h lives in LDS bf16 planes; Whh planes stream from L2 each step.
__global__ __launch_bounds__(1024) void k2_scan(
    const float* __restrict__ xw_all,
    const __bf16* __restrict__ WhhR, const __bf16* __restrict__ WhhI, const __bf16* __restrict__ WhhS,
    __bf16* __restrict__ hs_all) {
    __shared__ __attribute__((aligned(16))) __bf16 hR[32 * 256], hI[32 * 256], hS[32 * 256];
    const int inp = blockIdx.x;
    const float* xw = xw_all + (size_t)inp * BTH * 2;
    __bf16* hsR = hs_all + (size_t)inp * 3 * BTH;
    __bf16* hsI = hsR + BTH;
    __bf16* hsS = hsI + BTH;

    const int tid = threadIdx.x, lane = tid & 31, wv = tid >> 5;
    const int mt = wv >> 4, nt = wv & 15;
    const int grp = lane >> 4, col = lane & 15;
    const int hcol = nt * 16 + col;
    const int crow = tid >> 5, c8 = (tid & 31) * 8;      // cooperative LDS->global copy map

    for (int e = tid; e < 32 * 256; e += 1024) {
        hR[e] = (__bf16)0.f; hI[e] = (__bf16)0.f; hS[e] = (__bf16)0.f;
    }
    __syncthreads();

    for (int t = 0; t < NT; ++t) {
        v8f t1 = {0,0,0,0,0,0,0,0}, t2 = t1, t3 = t1;
#pragma unroll
        for (int kk = 0; kk < 256; kk += 32) {
            v16bf ar = fragA(hR, (size_t)mt * 16, lane, kk);
            v16bf ai = fragA(hI, (size_t)mt * 16, lane, kk);
            v16bf as = fragA(hS, (size_t)mt * 16, lane, kk);
            v16bf br = fragB(WhhR, nt * 16, lane, kk);
            v16bf bi = fragB(WhhI, nt * 16, lane, kk);
            v16bf bs = fragB(WhhS, nt * 16, lane, kk);
            t1 = bwmma(ar, br, t1); t2 = bwmma(ai, bi, t2); t3 = bwmma(as, bs, t3);
        }
        cplx nh[8];
#pragma unroll
        for (int r = 0; r < 8; ++r) {
            int m = mt * 16 + r + 8 * grp;
            const float* xp = xw + (((size_t)t * NB + m) * NH + hcol) * 2;
            nh[r] = ctanhz(t1[r] - t2[r] + xp[0], t3[r] - t1[r] - t2[r] + xp[1]);
        }
        __syncthreads();                                  // everyone done reading h(t-1)
#pragma unroll
        for (int r = 0; r < 8; ++r) {
            int m = mt * 16 + r + 8 * grp;
            int idx = m * 256 + hcol;
            hR[idx] = (__bf16)nh[r].x;
            hI[idx] = (__bf16)nh[r].y;
            hS[idx] = (__bf16)(nh[r].x + nh[r].y);
        }
        __syncthreads();                                  // h(t) visible to all
        size_t g = ((size_t)crow * NT + t) * 256 + c8;    // hs rows flattened (b,t)
        *(uv4*)(hsR + g) = *(const uv4*)(hR + crow * 256 + c8);
        *(uv4*)(hsI + g) = *(const uv4*)(hI + crow * 256 + c8);
        *(uv4*)(hsS + g) = *(const uv4*)(hS + crow * 256 + c8);
    }
}

// ---------------- k3: embeds = ctanh(hs * We^T + be), score = |embeds . Wl + bl| ----------------
__global__ __launch_bounds__(256) void k3_embed(
    const __bf16* __restrict__ hsR, const __bf16* __restrict__ hsI, const __bf16* __restrict__ hsS,
    const __bf16* __restrict__ WeR, const __bf16* __restrict__ WeI, const __bf16* __restrict__ WeS,
    const float* __restrict__ be, const float* __restrict__ Wl, const float* __restrict__ bl,
    float* __restrict__ emb, float* __restrict__ sc) {
    __shared__ __attribute__((aligned(16))) cplx etile[16 * 256];
    __shared__ cplx part[256];
    const int tid = threadIdx.x, lane = tid & 31, wv = tid >> 5;
    const size_t mrow0 = (size_t)blockIdx.x * 16;
    const int grp = lane >> 4, col = lane & 15;
    const int nt0 = wv * 2;

    v8f t1a = {0,0,0,0,0,0,0,0}, t2a = t1a, t3a = t1a, t1b = t1a, t2b = t1a, t3b = t1a;
#pragma unroll
    for (int kk = 0; kk < 256; kk += 32) {
        v16bf ar = fragA(hsR, mrow0, lane, kk);
        v16bf ai = fragA(hsI, mrow0, lane, kk);
        v16bf as = fragA(hsS, mrow0, lane, kk);
        v16bf b0r = fragB(WeR, nt0 * 16, lane, kk);
        v16bf b0i = fragB(WeI, nt0 * 16, lane, kk);
        v16bf b0s = fragB(WeS, nt0 * 16, lane, kk);
        t1a = bwmma(ar, b0r, t1a); t2a = bwmma(ai, b0i, t2a); t3a = bwmma(as, b0s, t3a);
        v16bf b1r = fragB(WeR, nt0 * 16 + 16, lane, kk);
        v16bf b1i = fragB(WeI, nt0 * 16 + 16, lane, kk);
        v16bf b1s = fragB(WeS, nt0 * 16 + 16, lane, kk);
        t1b = bwmma(ar, b1r, t1b); t2b = bwmma(ai, b1i, t2b); t3b = bwmma(as, b1s, t3b);
    }
#pragma unroll
    for (int s = 0; s < 2; ++s) {
        int i = (nt0 + s) * 16 + col;
        float ber = be[2 * i], bei = be[2 * i + 1];
        v8f T1 = s ? t1b : t1a;
        v8f T2 = s ? t2b : t2a;
        v8f T3 = s ? t3b : t3a;
#pragma unroll
        for (int r = 0; r < 8; ++r) {
            int ml = r + 8 * grp;
            cplx e = ctanhz(T1[r] - T2[r] + ber, T3[r] - T1[r] - T2[r] + bei);
            etile[ml * 256 + i] = e;
            ((cplx*)emb)[(mrow0 + ml) * NI + i] = e;
        }
    }
    __syncthreads();
    // per-row complex dot with Wl, then magnitude
    int rrow = tid >> 4, chunk = tid & 15;
    float sr = 0.f, si = 0.f;
#pragma unroll
    for (int j = 0; j < 16; ++j) {
        int i = chunk * 16 + j;
        cplx e = etile[rrow * 256 + i];
        float wr = Wl[2 * i], wi = Wl[2 * i + 1];
        sr += e.x * wr - e.y * wi;
        si += e.x * wi + e.y * wr;
    }
    part[rrow * 16 + chunk] = {sr, si};
    __syncthreads();
    if (tid < 16) {
        float ar = bl[0], ai = bl[1];
#pragma unroll
        for (int j = 0; j < 16; ++j) { ar += part[tid * 16 + j].x; ai += part[tid * 16 + j].y; }
        sc[mrow0 + tid] = sqrtf(ar * ar + ai * ai);
    }
}

// ---------------- k4: softmax over T, attention pool, L2 normalize ----------------
__global__ __launch_bounds__(256) void k4_pool(const float* __restrict__ emb_all,
                                               const float* __restrict__ sc_all,
                                               float* __restrict__ out) {
    __shared__ float attn[1024];
    __shared__ float red[256];
    const int tid = threadIdx.x;
    const int b = blockIdx.x, inp = blockIdx.y;
    const float* sc = sc_all + ((size_t)inp * NB + b) * NT;

    float mx = -3.0e38f;
    for (int t = tid; t < NT; t += 256) mx = fmaxf(mx, sc[t]);
    red[tid] = mx; __syncthreads();
    for (int s = 128; s > 0; s >>= 1) { if (tid < s) red[tid] = fmaxf(red[tid], red[tid + s]); __syncthreads(); }
    mx = red[0]; __syncthreads();

    float lsum = 0.f;
    for (int t = tid; t < NT; t += 256) { float e = __expf(sc[t] - mx); attn[t] = e; lsum += e; }
    red[tid] = lsum; __syncthreads();
    for (int s = 128; s > 0; s >>= 1) { if (tid < s) red[tid] += red[tid + s]; __syncthreads(); }
    const float inv = 1.f / red[0]; __syncthreads();

    const cplx* emb = (const cplx*)emb_all + ((size_t)inp * NB + b) * NT * NI;
    float ar = 0.f, ai = 0.f;
    const int i = tid;
    for (int t = 0; t < NT; ++t) {
        if (t + 8 < NT) __builtin_prefetch(&emb[(size_t)(t + 8) * NI + i], 0, 0);
        cplx e = emb[(size_t)t * NI + i];
        float a = attn[t] * inv;
        ar += e.x * a; ai += e.y * a;
    }
    red[tid] = ar * ar + ai * ai; __syncthreads();
    for (int s = 128; s > 0; s >>= 1) { if (tid < s) red[tid] += red[tid + s]; __syncthreads(); }
    const float invn = 1.f / sqrtf(red[0]);
    cplx* o = (cplx*)out + ((size_t)inp * NB + b) * NI;
    o[i] = {ar * invn, ai * invn};
}

// ---------------- host launcher ----------------
extern "C" void kernel_launch(void* const* d_in, const int* in_sizes, int n_in,
                              void* d_out, int out_size, void* d_ws, size_t ws_size,
                              hipStream_t stream) {
    const float* anchor   = (const float*)d_in[0];
    const float* positive = (const float*)d_in[1];
    const float* negative = (const float*)d_in[2];
    const float* Wih = (const float*)d_in[3];
    const float* Whh = (const float*)d_in[4];
    const float* bih = (const float*)d_in[5];
    const float* bhh = (const float*)d_in[6];
    const float* We  = (const float*)d_in[7];
    const float* be  = (const float*)d_in[8];
    const float* Wl  = (const float*)d_in[9];
    const float* bl  = (const float*)d_in[10];
    (void)in_sizes; (void)n_in; (void)out_size; (void)ws_size;

    char* ws = (char*)d_ws;
    __bf16* wplanes = (__bf16*)ws;
    float*  xw  = (float*)(ws + OFF_XW);
    __bf16* hs  = (__bf16*)(ws + OFF_HS);
    float*  emb = (float*)(ws + OFF_EMB);
    float*  sc  = (float*)(ws + OFF_SC);

    k_prep<<<768, 256, 0, stream>>>(Wih, Whh, We, wplanes);

    const float* xs[3] = {anchor, positive, negative};
    for (int i = 0; i < 3; ++i) {
        k1_inproj<<<2048, 256, 0, stream>>>(xs[i], bih, bhh,
            wplanes + 0 * PLANE_E, wplanes + 1 * PLANE_E, wplanes + 2 * PLANE_E,
            xw + (size_t)i * BTH * 2);
    }

    k2_scan<<<3, 1024, 0, stream>>>(xw,
        wplanes + 3 * PLANE_E, wplanes + 4 * PLANE_E, wplanes + 5 * PLANE_E, hs);

    for (int i = 0; i < 3; ++i) {
        __bf16* hsR = hs + (size_t)i * 3 * BTH;
        k3_embed<<<2048, 256, 0, stream>>>(hsR, hsR + BTH, hsR + 2 * BTH,
            wplanes + 6 * PLANE_E, wplanes + 7 * PLANE_E, wplanes + 8 * PLANE_E,
            be, Wl, bl,
            emb + (size_t)i * BT * NI * 2, sc + (size_t)i * BT);
    }

    k4_pool<<<dim3(NB, 3), 256, 0, stream>>>(emb, sc, (float*)d_out);
}
